// SelfAttention_69329362092508
// MI455X (gfx1250) — matrix operand order repaired and checked by reading, not compiled, not run on hardware
//
#include <hip/hip_runtime.h>
#include <math.h>

// Problem constants (match reference)
#define B_    8
#define CIN_  4160   // 130 * 32 -> exact K-loop
#define N_    1024   // H*W
#define C_    512

typedef __attribute__((ext_vector_type(16))) _Float16 v16h;
typedef __attribute__((ext_vector_type(8)))  _Float16 v8h;
typedef __attribute__((ext_vector_type(4)))  _Float16 v4h;
typedef __attribute__((ext_vector_type(8)))  float    v8f;
typedef __attribute__((ext_vector_type(4)))  float    v4f;
typedef __attribute__((ext_vector_type(4)))  unsigned int u32x4;
typedef __attribute__((ext_vector_type(4)))  int      i32x4;
typedef __attribute__((ext_vector_type(8)))  int      i32x8;

#ifndef __has_builtin
#define __has_builtin(x) 0
#endif
#if __has_builtin(__builtin_amdgcn_tensor_load_to_lds) && \
    __has_builtin(__builtin_amdgcn_s_wait_tensorcnt)
#define USE_TDM 1
#else
#define USE_TDM 0
#endif

// ---------------------------------------------------------------------------
// WMMA fragment loaders (wave32); A and B use the same contiguous k order:
//  A (16x32 f16): lane L -> row M=L%16, g=L/16; e -> K=e+16g (contiguous 16 halfs)
//  B (32x16 f16): lane L -> col N=L%16, g=L/16; e -> K=e+16g (contiguous 16 halfs)
//  D (16x16 f32): lane L -> col N=L%16; VGPR r -> M=r+8g
// Storage: row-major [rows][ld]; A rows==M, B rows==fragment-N index.
// ---------------------------------------------------------------------------
static __device__ __forceinline__ v16h load_a_frag(const _Float16* base, int ld,
                                                   int m0, int k0) {
  const int lane = threadIdx.x & 31;
  const int m = m0 + (lane & 15);
  const int g = lane >> 4;
  const _Float16* p = base + (size_t)m * ld + (k0 + 16 * g);
  v8h lo = *(const v8h*)(p);
  v8h hi = *(const v8h*)(p + 8);
  v16h r;
#pragma unroll
  for (int e = 0; e < 8; ++e) { r[e] = lo[e]; r[e + 8] = hi[e]; }
  return r;
}

static __device__ __forceinline__ v16h load_b_frag(const _Float16* base, int ld,
                                                   int n0, int k0) {
  const int lane = threadIdx.x & 31;
  const int n = n0 + (lane & 15);
  const int g = lane >> 4;
  return *(const v16h*)(base + (size_t)n * ld + (k0 + 16 * g));
}

static __device__ __forceinline__ v8f wmma_f16(v16h a, v16h b, v8f c) {
  return __builtin_amdgcn_wmma_f32_16x16x32_f16(false, a, false, b, (short)0, c,
                                                false, false);
}

#if USE_TDM
// ---------------------------------------------------------------------------
// Tensor Data Mover: 2D tile load global->LDS (D# per ISA ch.8).
// tile_d0 elements contiguous (dim0), tile_d1 rows, row stride = stride0
// elements, element size = 2 bytes. Groups 2/3 zero (<=2D tensor).
// ---------------------------------------------------------------------------
static __device__ __forceinline__ void
tdm_load_tile_f16(unsigned lds_byte_off, const _Float16* gptr,
                  unsigned tensor_d0, unsigned tensor_d1, unsigned stride0,
                  unsigned tile_d0, unsigned tile_d1) {
  unsigned long long ga = (unsigned long long)(size_t)gptr;
  u32x4 g0;
  g0[0] = 1u;                                   // count=1, user mode, no gather
  g0[1] = lds_byte_off;                         // [63:32]  lds_addr
  g0[2] = (unsigned)(ga & 0xFFFFFFFFu);         // [95:64]  global_addr lo
  g0[3] = (unsigned)((ga >> 32) & 0x01FFFFFFu)  // [120:96] global_addr hi
        | (2u << 30);                           // [127:126] type = 2 (image)
  i32x8 g1;
  g1[0] = (int)(1u << 16);                      // data_size=1 (2 bytes), mask=0
  g1[1] = (int)((tensor_d0 & 0xFFFFu) << 16);   // [79:48] tensor_dim0 (lo16)
  g1[2] = (int)((tensor_d0 >> 16) | ((tensor_d1 & 0xFFFFu) << 16));
  g1[3] = (int)((tensor_d1 >> 16) | (tile_d0 << 16));   // [127:112] tile_dim0
  g1[4] = (int)(tile_d1 & 0xFFFFu);             // [143:128] tile_dim1, tile_dim2=0
  g1[5] = (int)stride0;                         // [207:160] tensor_dim0_stride
  g1[6] = 0;
  g1[7] = 0;
  i32x4 z4 = {};
#if __clang_major__ >= 23
  i32x8 z8 = {};
  __builtin_amdgcn_tensor_load_to_lds(g0, g1, z4, z4, z8, 0);
#else
  __builtin_amdgcn_tensor_load_to_lds(g0, g1, z4, z4, 0);
#endif
}
#endif  // USE_TDM

// ---------------------------------------------------------------------------
// f32 -> f16 weight cast (x4 vectorized)
// ---------------------------------------------------------------------------
__global__ void cast_f32_f16(const float* __restrict__ src,
                             _Float16* __restrict__ dst, int n4) {
  int i = blockIdx.x * blockDim.x + threadIdx.x;
  if (i < n4) {
    v4f v = ((const v4f*)src)[i];
    v4h h;
#pragma unroll
    for (int e = 0; e < 4; ++e) h[e] = (_Float16)v[e];
    ((v4h*)dst)[i] = h;
  }
}

// ---------------------------------------------------------------------------
// Projection: out[c,i] = scale*(sum_cin W[c,cin]*x[cin,i] + bias[c])
// GEMM: M=i, Nmat=c, K=cin. Block tile 64i x 128c, 8 waves as 2(Mw) x 4(Nw),
// each wave 32x32 (2x2 WMMA tiles, 4 wmma per K-step).
//  A: x tile f32->f16 transposed into LDS by all 256 threads.
//  B: W tile staged by TDM (tensor_load_to_lds), double buffered; fragment
//     reads are contiguous 32B runs from LDS.
// nat_out=0 -> [b][i][c] f16 (q,k);  nat_out=1 -> [b][c][i] f16 (v)
// ---------------------------------------------------------------------------
__global__ void __launch_bounds__(256)
proj_kernel(const float* __restrict__ x, const _Float16* __restrict__ Wh,
            const float* __restrict__ bias, _Float16* __restrict__ out,
            float scale, int nat_out) {
  __shared__ _Float16 xs[64][32];        // [i-local][k]  (transposed x tile)
  __shared__ _Float16 wsb[2][128][32];   // [buf][c-local][k] (TDM target)

  const int b    = blockIdx.z;
  const int i0   = blockIdx.x * 64;
  const int cblk = blockIdx.y * 128;
  const int wave = threadIdx.x >> 5;
  const int lane = threadIdx.x & 31;
  const int iw   = (wave & 1) * 32;      // wave's i sub-tile
  const int cw   = (wave >> 1) * 32;     // wave's c sub-tile
  const int nl   = lane & 15;
  const int g    = lane >> 4;

  const float* xb = x + (size_t)b * CIN_ * N_;

  // x loader mapping: 2048 f32 per tile, 8 per thread
  const int kk  = threadIdx.x >> 3;      // 0..31
  const int seg = threadIdx.x & 7;       // ii = seg*8 .. seg*8+7

#if USE_TDM
  if (wave == 0)  // prologue: stage first W tile
    tdm_load_tile_f16((unsigned)(size_t)&wsb[0][0][0],
                      Wh + (size_t)cblk * CIN_, CIN_, C_, CIN_, 32, 128);
#endif

  v8f acc[2][2] = {};
  for (int s = 0, k0 = 0; k0 < CIN_; ++s, k0 += 32) {
    const int cur = s & 1;
    v4f xv0 = *(const v4f*)(xb + (size_t)(k0 + kk) * N_ + i0 + seg * 8);
    v4f xv1 = *(const v4f*)(xb + (size_t)(k0 + kk) * N_ + i0 + seg * 8 + 4);
    __syncthreads();  // previous iteration's readers of xs / wsb done
#if USE_TDM
    if (wave == 0) {
      if (k0 + 32 < CIN_) {   // issue next tile, then wait for current
        tdm_load_tile_f16((unsigned)(size_t)&wsb[cur ^ 1][0][0],
                          Wh + (size_t)cblk * CIN_ + k0 + 32, CIN_, C_, CIN_,
                          32, 128);
        __builtin_amdgcn_s_wait_tensorcnt(1);
      } else {
        __builtin_amdgcn_s_wait_tensorcnt(0);
      }
    }
#else
    {  // cooperative W tile copy: 128 rows x 32 halfs, 2 threads/row
      const int row = threadIdx.x >> 1, part = threadIdx.x & 1;
      *(v16h*)&wsb[cur][row][part * 16] =
          *(const v16h*)(Wh + (size_t)(cblk + row) * CIN_ + k0 + part * 16);
    }
#endif
#pragma unroll
    for (int q = 0; q < 4; ++q) {
      xs[seg * 8 + q][kk]     = (_Float16)xv0[q];
      xs[seg * 8 + 4 + q][kk] = (_Float16)xv1[q];
    }
    __syncthreads();  // xs + wsb[cur] visible

    v16h a0 = load_a_frag(&xs[0][0], 32, iw, 0);
    v16h a1 = load_a_frag(&xs[0][0], 32, iw + 16, 0);
    v16h b0 = load_b_frag(&wsb[cur][0][0], 32, cw, 0);
    v16h b1 = load_b_frag(&wsb[cur][0][0], 32, cw + 16, 0);
    acc[0][0] = wmma_f16(a0, b0, acc[0][0]);
    acc[0][1] = wmma_f16(a0, b1, acc[0][1]);
    acc[1][0] = wmma_f16(a1, b0, acc[1][0]);
    acc[1][1] = wmma_f16(a1, b1, acc[1][1]);
  }

#pragma unroll
  for (int mi = 0; mi < 2; ++mi)
#pragma unroll
    for (int nj = 0; nj < 2; ++nj) {
      const int   it = i0 + iw + mi * 16;
      const int   c  = cblk + cw + nj * 16 + nl;
      const float bv = bias[c];
      if (nat_out) {  // v[b][c][i]: 8 contiguous halfs per lane
        _Float16* dst = out + (size_t)b * C_ * N_ + (size_t)c * N_ + it + 8 * g;
        v8h h;
#pragma unroll
        for (int r = 0; r < 8; ++r) h[r] = (_Float16)((acc[mi][nj][r] + bv) * scale);
        *(v8h*)dst = h;
      } else {        // qT/kT[b][i][c]
        _Float16* dst = out + (size_t)b * N_ * C_;
#pragma unroll
        for (int r = 0; r < 8; ++r)
          dst[(size_t)(it + 8 * g + r) * C_ + c] =
              (_Float16)((acc[mi][nj][r] + bv) * scale);
      }
    }
}

// ---------------------------------------------------------------------------
// Scores: S[i,j] = sum_c qT[i,c]*kT[j,c]. Block 128i x 64j, wave 32x32.
// ---------------------------------------------------------------------------
__global__ void __launch_bounds__(256)
scores_kernel(const _Float16* __restrict__ qT, const _Float16* __restrict__ kT,
              float* __restrict__ S) {
  const int b    = blockIdx.z;
  const int wave = threadIdx.x >> 5;
  const int lane = threadIdx.x & 31;
  const int ib   = blockIdx.x * 128 + (wave & 3) * 32;
  const int jb   = blockIdx.y * 64 + (wave >> 2) * 32;
  const _Float16* qb = qT + (size_t)b * N_ * C_;
  const _Float16* kb = kT + (size_t)b * N_ * C_;

  v8f acc[2][2] = {};
  for (int k0 = 0; k0 < C_; k0 += 32) {
    v16h a0 = load_a_frag(qb, C_, ib, k0);
    v16h a1 = load_a_frag(qb, C_, ib + 16, k0);
    v16h b0 = load_b_frag(kb, C_, jb, k0);
    v16h b1 = load_b_frag(kb, C_, jb + 16, k0);
    acc[0][0] = wmma_f16(a0, b0, acc[0][0]);
    acc[0][1] = wmma_f16(a0, b1, acc[0][1]);
    acc[1][0] = wmma_f16(a1, b0, acc[1][0]);
    acc[1][1] = wmma_f16(a1, b1, acc[1][1]);
  }
  const int n = lane & 15, g = lane >> 4;
  float* Sb = S + (size_t)b * N_ * N_;
#pragma unroll
  for (int mi = 0; mi < 2; ++mi)
#pragma unroll
    for (int nj = 0; nj < 2; ++nj)
#pragma unroll
      for (int r = 0; r < 8; ++r)
        Sb[(size_t)(ib + mi * 16 + 8 * g + r) * N_ + jb + nj * 16 + n] =
            acc[mi][nj][r];
}

// ---------------------------------------------------------------------------
// Row softmax over N_=1024 scores -> f16 probabilities.
// ---------------------------------------------------------------------------
__global__ void __launch_bounds__(256)
softmax_kernel(const float* __restrict__ S, _Float16* __restrict__ P) {
  const int row = blockIdx.x;
  const int tid = threadIdx.x;
  const float* s = S + (size_t)row * N_;
  v4f v = ((const v4f*)s)[tid];

  __shared__ float red[256];
  red[tid] = fmaxf(fmaxf(v[0], v[1]), fmaxf(v[2], v[3]));
  __syncthreads();
  for (int st = 128; st > 0; st >>= 1) {
    if (tid < st) red[tid] = fmaxf(red[tid], red[tid + st]);
    __syncthreads();
  }
  const float rowmax = red[0];
  __syncthreads();

  float e0 = __expf(v[0] - rowmax), e1 = __expf(v[1] - rowmax);
  float e2 = __expf(v[2] - rowmax), e3 = __expf(v[3] - rowmax);
  red[tid] = e0 + e1 + e2 + e3;
  __syncthreads();
  for (int st = 128; st > 0; st >>= 1) {
    if (tid < st) red[tid] += red[tid + st];
    __syncthreads();
  }
  const float inv = 1.0f / red[0];

  v4h h;
  h[0] = (_Float16)(e0 * inv); h[1] = (_Float16)(e1 * inv);
  h[2] = (_Float16)(e2 * inv); h[3] = (_Float16)(e3 * inv);
  ((v4h*)(P + (size_t)row * N_))[tid] = h;
}

// ---------------------------------------------------------------------------
// AV: hT[i,c] = sum_j P[i,j]*v[c,j]. Block 128i x 64c, wave 32x32.
// ---------------------------------------------------------------------------
__global__ void __launch_bounds__(256)
av_kernel(const _Float16* __restrict__ P, const _Float16* __restrict__ Vn,
          _Float16* __restrict__ hT) {
  const int b    = blockIdx.z;
  const int wave = threadIdx.x >> 5;
  const int lane = threadIdx.x & 31;
  const int ib   = blockIdx.x * 128 + (wave & 3) * 32;
  const int cb   = blockIdx.y * 64 + (wave >> 2) * 32;
  const _Float16* Pb = P + (size_t)b * N_ * N_;
  const _Float16* Vb = Vn + (size_t)b * C_ * N_;

  v8f acc[2][2] = {};
  for (int j0 = 0; j0 < N_; j0 += 32) {
    v16h a0 = load_a_frag(Pb, N_, ib, j0);
    v16h a1 = load_a_frag(Pb, N_, ib + 16, j0);
    v16h b0 = load_b_frag(Vb, N_, cb, j0);
    v16h b1 = load_b_frag(Vb, N_, cb + 16, j0);
    acc[0][0] = wmma_f16(a0, b0, acc[0][0]);
    acc[0][1] = wmma_f16(a0, b1, acc[0][1]);
    acc[1][0] = wmma_f16(a1, b0, acc[1][0]);
    acc[1][1] = wmma_f16(a1, b1, acc[1][1]);
  }
  const int n = lane & 15, g = lane >> 4;
  _Float16* dst = hT + (size_t)b * N_ * C_;
#pragma unroll
  for (int mi = 0; mi < 2; ++mi)
#pragma unroll
    for (int nj = 0; nj < 2; ++nj)
#pragma unroll
      for (int r = 0; r < 8; ++r)
        dst[(size_t)(ib + mi * 16 + 8 * g + r) * C_ + cb + nj * 16 + n] =
            (_Float16)acc[mi][nj][r];
}

// ---------------------------------------------------------------------------
// Output projection: out[co,i] = sum_c Wo[co,c]*hT[i,c] + bo[co] (f32 out)
// Block 128co x 64i, wave 32x32.
// ---------------------------------------------------------------------------
__global__ void __launch_bounds__(256)
oproj_kernel(const _Float16* __restrict__ Woh, const float* __restrict__ bo,
             const _Float16* __restrict__ hT, float* __restrict__ out) {
  const int b    = blockIdx.z;
  const int wave = threadIdx.x >> 5;
  const int lane = threadIdx.x & 31;
  const int cob  = blockIdx.x * 128 + (wave & 3) * 32;
  const int ib   = blockIdx.y * 64 + (wave >> 2) * 32;
  const _Float16* hb = hT + (size_t)b * N_ * C_;

  v8f acc[2][2] = {};
  for (int k0 = 0; k0 < C_; k0 += 32) {
    v16h a0 = load_a_frag(Woh, C_, cob, k0);
    v16h a1 = load_a_frag(Woh, C_, cob + 16, k0);
    v16h b0 = load_b_frag(hb, C_, ib, k0);
    v16h b1 = load_b_frag(hb, C_, ib + 16, k0);
    acc[0][0] = wmma_f16(a0, b0, acc[0][0]);
    acc[0][1] = wmma_f16(a0, b1, acc[0][1]);
    acc[1][0] = wmma_f16(a1, b0, acc[1][0]);
    acc[1][1] = wmma_f16(a1, b1, acc[1][1]);
  }
  const int n = lane & 15, g = lane >> 4;
  float* ob = out + (size_t)b * C_ * N_;
#pragma unroll
  for (int mi = 0; mi < 2; ++mi)
#pragma unroll
    for (int nj = 0; nj < 2; ++nj)
#pragma unroll
      for (int r = 0; r < 8; ++r) {
        const int co = cob + mi * 16 + 8 * g + r;
        ob[(size_t)co * N_ + ib + nj * 16 + n] = acc[mi][nj][r] + bo[co];
      }
}

// ---------------------------------------------------------------------------
extern "C" void kernel_launch(void* const* d_in, const int* in_sizes, int n_in,
                              void* d_out, int out_size, void* d_ws,
                              size_t ws_size, hipStream_t stream) {
  (void)in_sizes; (void)n_in; (void)out_size; (void)ws_size;

  const float* x1 = (const float*)d_in[0];
  const float* x2 = (const float*)d_in[1];
  const float* Wq = (const float*)d_in[2];
  const float* bq = (const float*)d_in[3];
  const float* Wk = (const float*)d_in[4];
  const float* bk = (const float*)d_in[5];
  const float* Wv = (const float*)d_in[6];
  const float* bv = (const float*)d_in[7];
  const float* Wo = (const float*)d_in[8];
  const float* bo = (const float*)d_in[9];
  float* out = (float*)d_out;

  char* ws = (char*)d_ws;
  size_t off = 0;
  auto take = [&](size_t bytes) -> char* {
    char* p = ws + off;
    off += (bytes + 255) & ~(size_t)255;
    return p;
  };

  _Float16* Wqh = (_Float16*)take((size_t)C_ * CIN_ * 2);
  _Float16* Wkh = (_Float16*)take((size_t)C_ * CIN_ * 2);
  _Float16* Wvh = (_Float16*)take((size_t)C_ * CIN_ * 2);
  _Float16* Woh = (_Float16*)take((size_t)C_ * C_ * 2);
  _Float16* qT  = (_Float16*)take((size_t)B_ * N_ * C_ * 2);   // [b][i][c]
  _Float16* kT  = (_Float16*)take((size_t)B_ * N_ * C_ * 2);   // [b][j][c]
  _Float16* Vn  = (_Float16*)take((size_t)B_ * C_ * N_ * 2);   // [b][c][j]
  float*    S   = (float*)   take((size_t)B_ * N_ * N_ * 4);   // [b][i][j]
  _Float16* Pp  = (_Float16*)take((size_t)B_ * N_ * N_ * 2);   // softmax(S)
  _Float16* hT  = (_Float16*)take((size_t)B_ * N_ * C_ * 2);   // [b][i][c]

  {  // weight casts
    int n4 = C_ * CIN_ / 4;
    cast_f32_f16<<<(n4 + 255) / 256, 256, 0, stream>>>(Wq, Wqh, n4);
    cast_f32_f16<<<(n4 + 255) / 256, 256, 0, stream>>>(Wk, Wkh, n4);
    cast_f32_f16<<<(n4 + 255) / 256, 256, 0, stream>>>(Wv, Wvh, n4);
    int m4 = C_ * C_ / 4;
    cast_f32_f16<<<(m4 + 255) / 256, 256, 0, stream>>>(Wo, Woh, m4);
  }

  // QKV projections (softmax scale folded into q)
  const float qscale = 1.0f / sqrtf((float)C_);
  dim3 gp(N_ / 64, C_ / 128, B_);
  proj_kernel<<<gp, 256, 0, stream>>>(x1, Wqh, bq, qT, qscale, 0);
  proj_kernel<<<gp, 256, 0, stream>>>(x2, Wkh, bk, kT, 1.0f, 0);
  proj_kernel<<<gp, 256, 0, stream>>>(x2, Wvh, bv, Vn, 1.0f, 1);

  dim3 gs(N_ / 128, N_ / 64, B_);
  scores_kernel<<<gs, 256, 0, stream>>>(qT, kT, S);

  softmax_kernel<<<B_ * N_, 256, 0, stream>>>(S, Pp);

  dim3 ga(N_ / 128, C_ / 64, B_);
  av_kernel<<<ga, 256, 0, stream>>>(Pp, Vn, hT);

  dim3 go(C_ / 128, N_ / 64, B_);
  oproj_kernel<<<go, 256, 0, stream>>>(Woh, bo, hT, out);
}
